// SpatialAttentionLayer_70145405878954
// MI455X (gfx1250) — matrix operand order repaired and checked by reading, not compiled, not run on hardware
//
#include <hip/hip_runtime.h>
#include <cstdint>

typedef __bf16 bf16_t;
typedef __attribute__((ext_vector_type(16))) __bf16 v16bf;
typedef __attribute__((ext_vector_type(8)))  __bf16 v8bf;
typedef __attribute__((ext_vector_type(4)))  __bf16 v4bf;
typedef __attribute__((ext_vector_type(8)))  float  v8f;
typedef __attribute__((ext_vector_type(4)))  float  v4f;

// Problem constants
#define TT  128
#define BBb 64
#define NNj 24
#define DDd 256
#define HHh 8
#define FFf 32
#define TBP (TT*BBb)        // 8192 (t,b) positions
#define TBN (TT*BBb*NNj)    // 196608 rows of x

__device__ __forceinline__ v16bf cat16(v8bf lo, v8bf hi) {
  return __builtin_shufflevector(lo, hi, 0,1,2,3,4,5,6,7,8,9,10,11,12,13,14,15);
}
__device__ __forceinline__ v8f wmma_bf16(v16bf a, v16bf b, v8f c) {
  // D = A(16x32 bf16) * B(32x16 bf16) + C(16x16 f32)
  return __builtin_amdgcn_wmma_f32_16x16x32_bf16(false, a, false, b, (short)0, c, false, false);
}

// Async global->LDS copy of 16B per lane (GLOBAL_LOAD_ASYNC_TO_LDS_B128, GVS
// mode: mem = saddr + vaddr). ldsb = per-lane LDS byte address (low 32 bits of
// the flat shared address), gvo = per-lane byte offset against 'base'.
__device__ __forceinline__ void async_b128(unsigned ldsb, unsigned gvo, const void* base) {
  asm volatile("global_load_async_to_lds_b128 %0, %1, %2"
               :: "v"(ldsb), "v"(gvo), "s"(base) : "memory");
}
__device__ __forceinline__ void wait_async0() {
  asm volatile("s_wait_asynccnt 0x0" ::: "memory");
}

// ---------------------------------------------------------------------------
// Kernel 1: weight conversion fp32 -> bf16 (+ repack Wq (H,N,F,D) -> (N,HF,D))
// ---------------------------------------------------------------------------
__global__ __launch_bounds__(256) void cvt_weights(
    const float* __restrict__ Wk, const float* __restrict__ Wv,
    const float* __restrict__ Wq,
    bf16_t* __restrict__ WkB, bf16_t* __restrict__ WvB, bf16_t* __restrict__ WqB)
{
  const int stride = gridDim.x * blockDim.x;
  const int i0 = blockIdx.x * blockDim.x + threadIdx.x;
  for (int i = i0; i < HHh*FFf*DDd; i += stride) {       // 65536
    WkB[i] = (bf16_t)Wk[i];
    WvB[i] = (bf16_t)Wv[i];
  }
  for (int e = i0; e < HHh*NNj*FFf*DDd; e += stride) {   // 1572864
    const int d = e & 255;
    const int f = (e >> 8) & 31;
    const int t = e >> 13;          // h*24 + n
    const int n = t % 24;
    const int h = t / 24;
    WqB[((n * 256 + h * 32 + f) << 8) + d] = (bf16_t)Wq[e];
  }
}

// ---------------------------------------------------------------------------
// Kernel 2: K/V GEMM.  rows r=(tb,joint) of x (196608 x 256) @ W^T (256 x 256)
// grid.x = row tile (64 rows), grid.y = 0:K 1:V
// K stored as (tb,h,joint[32],f)  (B-matrix for scores)
// V stored as (tb,h,f,joint[32])  (B-matrix for out-GEMM; pad joints pre-zeroed)
// B weight tiles staged via GLOBAL_LOAD_ASYNC_TO_LDS_B128 (ASYNCcnt).
// ---------------------------------------------------------------------------
__global__ __launch_bounds__(256) void kv_gemm(
    const float* __restrict__ x,
    const bf16_t* __restrict__ WkB, const bf16_t* __restrict__ WvB,
    const float* __restrict__ bk,  const float* __restrict__ bv,
    bf16_t* __restrict__ wsK, bf16_t* __restrict__ wsV)
{
  __shared__ bf16_t Ash[64 * 264];   // 64 rows x 256 k (stride 264) bf16
  __shared__ bf16_t Bsh[256 * 32];   // 256 cols x 32 k  bf16 (col-major tiles)

  const int mode = blockIdx.y;
  const bf16_t* __restrict__ W    = mode ? WvB : WkB;
  const float*  __restrict__ bias = mode ? bv  : bk;

  const int tid  = threadIdx.x;
  const int lane = tid & 31;
  const int wave = tid >> 5;
  const int mw = wave & 3;           // 4 m-tiles of 16 rows
  const int cn = wave >> 2;          // 2 col halves of 128
  const int rBase = blockIdx.x * 64;
  const int l15 = lane & 15;
  const int hiHalf = (lane >= 16) ? 1 : 0;

  // per-lane async-copy geometry: wave moves cols [wave*32, wave*32+32)
  const unsigned BshBase = (unsigned)(uintptr_t)(void*)&Bsh[0];
  const unsigned colA = wave * 32 + (lane >> 2);   // col handled by this lane (+q*8)
  const unsigned seg  = (lane & 3) * 16;           // 16B segment within 64B column

  // ---- stage A: 64x256 fp32 -> bf16 LDS
  {
    const float* xb = x + (size_t)rBase * 256;
    const int row = tid >> 2;
    const int c0  = (tid & 3) * 64;
    #pragma unroll
    for (int c = 0; c < 16; ++c) {
      v4f fv = *(const v4f*)(xb + row * 256 + c0 + c * 4);
      v4bf b4;
      b4.x = (bf16_t)fv.x; b4.y = (bf16_t)fv.y;
      b4.z = (bf16_t)fv.z; b4.w = (bf16_t)fv.w;
      *(v4bf*)&Ash[row * 264 + c0 + c * 4] = b4;
    }
  }

  v8f acc[8] = {};

  for (int kk = 0; kk < 8; ++kk) {
    __syncthreads();                 // previous tile's consumers are done
    // ---- async-stage B tile: 256 cols x 32 k (64 B / column)
    {
      __builtin_prefetch(W + colA * 256 + (kk + 1) * 32, 0, 1);  // global_prefetch_b8
      #pragma unroll
      for (int q = 0; q < 4; ++q) {
        const unsigned col = colA + q * 8;
        async_b128(BshBase + col * 64 + seg,
                   col * 512 + kk * 64 + seg, W);
      }
      wait_async0();                 // my 512B x4 landed in LDS
    }
    __syncthreads();                 // all waves' segments visible

    // A fragment (ISA 16x32 bf16 layout)
    const bf16_t* ap = &Ash[(mw * 16 + l15) * 264 + kk * 32 + (hiHalf ? 8 : 0)];
    const v16bf a = cat16(*(const v8bf*)ap, *(const v8bf*)(ap + 16));

    #pragma unroll
    for (int nt = 0; nt < 8; ++nt) {
      const int col = cn * 128 + nt * 16 + l15;
      const bf16_t* bp = &Bsh[col * 32 + (hiHalf ? 8 : 0)];
      const v16bf b = cat16(*(const v8bf*)bp, *(const v8bf*)(bp + 16));
      acc[nt] = wmma_bf16(a, b, acc[nt]);
    }
  }

  // ---- epilogue: bias add, bf16 convert, layout-aware scatter
  #pragma unroll
  for (int nt = 0; nt < 8; ++nt) {
    const int col = cn * 128 + nt * 16 + l15;   // hf
    const float bb = bias[col];
    const int h = col >> 5;
    const int f = col & 31;
    #pragma unroll
    for (int j = 0; j < 8; ++j) {
      const int r  = rBase + mw * 16 + j + (hiHalf ? 8 : 0);
      const int tb = r / 24;
      const int jn = r - tb * 24;
      const bf16_t o = (bf16_t)(acc[nt][j] + bb);
      if (mode == 0)
        wsK[((((size_t)tb * 8 + h) * 32 + jn) << 5) + f] = o;
      else
        wsV[((((size_t)tb * 8 + h) * 32 + f) << 5) + jn] = o;
    }
  }
}

// ---------------------------------------------------------------------------
// Kernel 3: Q GEMM.  grid.y = joint n; rows = 64 (t,b) positions.
// Q stored as (tb,h,joint[32],f)  (A-matrix for scores)
// ---------------------------------------------------------------------------
__global__ __launch_bounds__(256) void q_gemm(
    const float* __restrict__ x,
    const bf16_t* __restrict__ WqB,
    const float* __restrict__ bq,
    bf16_t* __restrict__ wsQ)
{
  __shared__ bf16_t Ash[64 * 264];
  __shared__ bf16_t Bsh[256 * 32];

  const int n = blockIdx.y;
  const int tbBase = blockIdx.x * 64;
  const bf16_t* __restrict__ W    = WqB + ((size_t)n << 16);  // n*256*256
  const float*  __restrict__ bias = bq + n * 256;

  const int tid  = threadIdx.x;
  const int lane = tid & 31;
  const int wave = tid >> 5;
  const int mw = wave & 3;
  const int cn = wave >> 2;
  const int l15 = lane & 15;
  const int hiHalf = (lane >= 16) ? 1 : 0;

  const unsigned BshBase = (unsigned)(uintptr_t)(void*)&Bsh[0];
  const unsigned colA = wave * 32 + (lane >> 2);
  const unsigned seg  = (lane & 3) * 16;

  // ---- stage A: 64 tb-rows of joint n
  {
    const int row = tid >> 2;
    const int c0  = (tid & 3) * 64;
    const size_t gro = ((size_t)(tbBase + row) * 24 + n) * 256;
    #pragma unroll
    for (int c = 0; c < 16; ++c) {
      v4f fv = *(const v4f*)(x + gro + c0 + c * 4);
      v4bf b4;
      b4.x = (bf16_t)fv.x; b4.y = (bf16_t)fv.y;
      b4.z = (bf16_t)fv.z; b4.w = (bf16_t)fv.w;
      *(v4bf*)&Ash[row * 264 + c0 + c * 4] = b4;
    }
  }

  v8f acc[8] = {};

  for (int kk = 0; kk < 8; ++kk) {
    __syncthreads();
    {
      __builtin_prefetch(W + colA * 256 + (kk + 1) * 32, 0, 1);
      #pragma unroll
      for (int q = 0; q < 4; ++q) {
        const unsigned col = colA + q * 8;
        async_b128(BshBase + col * 64 + seg,
                   col * 512 + kk * 64 + seg, W);
      }
      wait_async0();
    }
    __syncthreads();

    const bf16_t* ap = &Ash[(mw * 16 + l15) * 264 + kk * 32 + (hiHalf ? 8 : 0)];
    const v16bf a = cat16(*(const v8bf*)ap, *(const v8bf*)(ap + 16));

    #pragma unroll
    for (int nt = 0; nt < 8; ++nt) {
      const int col = cn * 128 + nt * 16 + l15;
      const bf16_t* bp = &Bsh[col * 32 + (hiHalf ? 8 : 0)];
      const v16bf b = cat16(*(const v8bf*)bp, *(const v8bf*)(bp + 16));
      acc[nt] = wmma_bf16(a, b, acc[nt]);
    }
  }

  #pragma unroll
  for (int nt = 0; nt < 8; ++nt) {
    const int col = cn * 128 + nt * 16 + l15;
    const float bb = bias[col];
    const int h = col >> 5;
    const int f = col & 31;
    #pragma unroll
    for (int j = 0; j < 8; ++j) {
      const int tb = tbBase + mw * 16 + j + (hiHalf ? 8 : 0);
      const bf16_t o = (bf16_t)(acc[nt][j] + bb);
      wsQ[((((size_t)tb * 8 + h) * 32 + n) << 5) + f] = o;
    }
  }
}

// ---------------------------------------------------------------------------
// Kernel 4: attention. One wave per (tb, h). Fragments load straight from the
// WMMA-layout workspaces; softmax in LDS over the 24 real joints.
// ---------------------------------------------------------------------------
__global__ __launch_bounds__(256) void attn_kernel(
    const bf16_t* __restrict__ wsQ, const bf16_t* __restrict__ wsK,
    const bf16_t* __restrict__ wsV, float* __restrict__ out)
{
  __shared__ float  Ssh[8][32 * 33];   // scores per wave
  __shared__ bf16_t Psh[8][32 * 32];   // attn weights per wave (A-matrix layout)

  const int tid  = threadIdx.x;
  const int lane = tid & 31;
  const int wave = tid >> 5;
  const int w  = blockIdx.x * 8 + wave;
  const int tb = w >> 3;
  const int h  = w & 7;
  const size_t base = (((size_t)tb * 8 + h) << 10);  // *32*32 elements

  const bf16_t* Qp = wsQ + base;   // (joint, f)
  const bf16_t* Kp = wsK + base;   // (joint, f)
  const bf16_t* Vp = wsV + base;   // (f, joint)

  const int l15 = lane & 15;
  const int hiHalf = (lane >= 16) ? 1 : 0;

  // scores = q @ k^T   (A = q 24x32, B = k^T 32x24; pad rows/cols ignored)
  v16bf aq[2], bkf[2];
  #pragma unroll
  for (int mt = 0; mt < 2; ++mt) {
    const bf16_t* p = Qp + (mt * 16 + l15) * 32 + (hiHalf ? 8 : 0);
    aq[mt] = cat16(*(const v8bf*)p, *(const v8bf*)(p + 16));
  }
  #pragma unroll
  for (int nt = 0; nt < 2; ++nt) {
    const bf16_t* p = Kp + (nt * 16 + l15) * 32 + (hiHalf ? 8 : 0);
    bkf[nt] = cat16(*(const v8bf*)p, *(const v8bf*)(p + 16));
  }
  #pragma unroll
  for (int mt = 0; mt < 2; ++mt) {
    #pragma unroll
    for (int nt = 0; nt < 2; ++nt) {
      v8f z = {};
      v8f s = wmma_bf16(aq[mt], bkf[nt], z);
      #pragma unroll
      for (int j = 0; j < 8; ++j) {
        const int row = mt * 16 + j + (hiHalf ? 8 : 0);
        const int col = nt * 16 + l15;
        Ssh[wave][row * 33 + col] = s[j];
      }
    }
  }
  __syncthreads();

  // softmax over the 24 real key-joints; lane < 24 handles one query row
  if (lane < 24) {
    const float scale = 0.1767766952966369f;   // 1/sqrt(F=32)
    float sv[24];
    float mx = -3.0e38f;
    #pragma unroll
    for (int m = 0; m < 24; ++m) {
      const float v = Ssh[wave][lane * 33 + m] * scale;
      sv[m] = v;
      mx = fmaxf(mx, v);
    }
    float sum = 0.f;
    #pragma unroll
    for (int m = 0; m < 24; ++m) {
      const float e = __expf(sv[m] - mx);
      sv[m] = e;
      sum += e;
    }
    const float inv = 1.0f / sum;
    #pragma unroll
    for (int m = 0; m < 24; ++m)
      Psh[wave][lane * 32 + m] = (bf16_t)(sv[m] * inv);
    #pragma unroll
    for (int m = 24; m < 32; ++m)
      Psh[wave][lane * 32 + m] = (bf16_t)0.0f;   // zero the K-pad
  }
  __syncthreads();

  // out = P @ V   (A = P 24x32(padded), B = V 32x32 with joints 24..31 == 0)
  v16bf ap[2], bvf[2];
  #pragma unroll
  for (int mt = 0; mt < 2; ++mt) {
    const bf16_t* p = &Psh[wave][(mt * 16 + l15) * 32 + (hiHalf ? 8 : 0)];
    ap[mt] = cat16(*(const v8bf*)p, *(const v8bf*)(p + 16));
  }
  #pragma unroll
  for (int ft = 0; ft < 2; ++ft) {
    const bf16_t* p = Vp + (ft * 16 + l15) * 32 + (hiHalf ? 8 : 0);
    bvf[ft] = cat16(*(const v8bf*)p, *(const v8bf*)(p + 16));
  }
  #pragma unroll
  for (int mt = 0; mt < 2; ++mt) {
    #pragma unroll
    for (int ft = 0; ft < 2; ++ft) {
      v8f z = {};
      v8f o = wmma_bf16(ap[mt], bvf[ft], z);
      #pragma unroll
      for (int j = 0; j < 8; ++j) {
        const int row = mt * 16 + j + (hiHalf ? 8 : 0);
        if (row < 24)
          out[(((size_t)tb * 24 + row) * 8 + h) * 32 + ft * 16 + l15] = o[j];
      }
    }
  }
}

// ---------------------------------------------------------------------------
extern "C" void kernel_launch(void* const* d_in, const int* in_sizes, int n_in,
                              void* d_out, int out_size, void* d_ws, size_t ws_size,
                              hipStream_t stream)
{
  (void)in_sizes; (void)n_in; (void)out_size; (void)ws_size;
  const float* x  = (const float*)d_in[0];
  const float* Wk = (const float*)d_in[1];
  const float* bk = (const float*)d_in[2];
  const float* Wv = (const float*)d_in[3];
  const float* bv = (const float*)d_in[4];
  const float* Wq = (const float*)d_in[5];
  const float* bq = (const float*)d_in[6];
  float* out = (float*)d_out;

  char* ws = (char*)d_ws;
  const size_t SZ = (size_t)TBP * 8 * 32 * 32 * 2;   // 128 MiB per tensor
  bf16_t* wsK = (bf16_t*)(ws);
  bf16_t* wsV = (bf16_t*)(ws + SZ);
  bf16_t* wsQ = (bf16_t*)(ws + 2 * SZ);
  bf16_t* WkB = (bf16_t*)(ws + 3 * SZ);
  bf16_t* WvB = WkB + 65536;
  bf16_t* WqB = WvB + 65536;

  // zero V workspace so the joint-pad rows (24..31) contract to 0 in WMMA
  hipMemsetAsync(wsV, 0, SZ, stream);

  cvt_weights<<<1024, 256, 0, stream>>>(Wk, Wv, Wq, WkB, WvB, WqB);
  kv_gemm<<<dim3(TBN / 64, 2), 256, 0, stream>>>(x, WkB, WvB, bk, bv, wsK, wsV);
  q_gemm<<<dim3(TBP / 64, NNj), 256, 0, stream>>>(x, WqB, bq, wsQ);
  attn_kernel<<<TBP, 256, 0, stream>>>(wsQ, wsK, wsV, out);
}
